// GANOKernel_32865089749352
// MI455X (gfx1250) — matrix operand order repaired and checked by reading, not compiled, not run on hardware
//
#include <hip/hip_runtime.h>
#include <hip/hip_bf16.h>
#include <math.h>

typedef __attribute__((ext_vector_type(16))) _Float16 v16h;
typedef __attribute__((ext_vector_type(8)))  _Float16 v8h;
typedef __attribute__((ext_vector_type(8)))  float    v8f;
typedef __attribute__((ext_vector_type(2)))  float    v2f;

#define NQ 2048
#define NO 2048
#define LAT 64
#define QB 16            // queries per block (one WMMA M tile)
#define WAVES 4
#define CHUNK 32         // obs per WMMA K step
#define OBS_PER_WAVE (NO / WAVES)          // 512
#define NCHUNK (OBS_PER_WAVE / CHUNK)      // 16
#define WPAD 68          // padded w_s row (17 banks stride -> <=2-way conflict)
#define RADIUS2 0.25f
#define NEGINF (-__builtin_inff())

#define FOREACH_Q(X) X(0) X(1) X(2) X(3) X(4) X(5) X(6) X(7) \
                     X(8) X(9) X(10) X(11) X(12) X(13) X(14) X(15)

// ---------------------------------------------------------------------------
// Prep 1: u[q][l] = pos_q . (W1[0:3]+W1[6:9]),  w[o][l] = pos_o . (W1[3:6]-W1[6:9]) + b1
// ---------------------------------------------------------------------------
__global__ void gano_prep_uw(const float* __restrict__ posq, const float* __restrict__ poso,
                             const float* __restrict__ W1, const float* __restrict__ b1,
                             float* __restrict__ u, float* __restrict__ w) {
  int idx = blockIdx.x * blockDim.x + threadIdx.x;
  const int n = NQ * LAT;
  if (idx < n) {
    int i = idx >> 6, l = idx & 63;
    const float* p = posq + i * 3;
    float a = 0.f;
#pragma unroll
    for (int d = 0; d < 3; ++d)
      a = fmaf(p[d], W1[d * LAT + l] + W1[(6 + d) * LAT + l], a);
    u[idx] = a;
  } else if (idx < 2 * n) {
    idx -= n;
    int i = idx >> 6, l = idx & 63;
    const float* p = poso + i * 3;
    float a = b1[l];
#pragma unroll
    for (int d = 0; d < 3; ++d)
      a = fmaf(p[d], W1[(3 + d) * LAT + l] - W1[(6 + d) * LAT + l], a);
    w[idx] = a;
  }
}

// ---------------------------------------------------------------------------
// Prep 2: v = h_obs @ Wv + bv, stored TRANSPOSED in f16: vT[l][o]  (B-matrix feed)
// ---------------------------------------------------------------------------
__global__ void gano_prep_v(const float* __restrict__ h_obs, const float* __restrict__ Wv,
                            const float* __restrict__ bv, _Float16* __restrict__ vT) {
  int idx = blockIdx.x * blockDim.x + threadIdx.x;
  if (idx >= NO * LAT) return;
  int o = idx >> 6, l = idx & 63;
  float a = bv[l];
#pragma unroll 8
  for (int k = 0; k < LAT; ++k)
    a = fmaf(h_obs[o * LAT + k], Wv[k * LAT + l], a);
  vT[l * NO + o] = (_Float16)a;
}

// ---------------------------------------------------------------------------
// Main: per-block 16 queries, 4 waves split the 2048 obs (fully decoupled in
// the main loop), flash-softmax + 5x v_wmma_f32_16x16x32_f16 per chunk:
// 4 tiles of P@V plus one P@ones tile accumulating the softmax denominator on
// the matrix pipe. Logits use packed-f32 (v_pk_add/v_pk_fma) VALU. 4-way merge.
// ---------------------------------------------------------------------------
__launch_bounds__(128, 1)
__global__ void gano_attn(const float* __restrict__ u, const float* __restrict__ w,
                          const _Float16* __restrict__ vT,
                          const float* __restrict__ posq, const float* __restrict__ poso,
                          const float* __restrict__ W2, const float* __restrict__ b2,
                          float* __restrict__ out) {
  __shared__ __align__(16) float u_s[QB][LAT];             // 4 KB
  __shared__ float posq_s[QB][4];
  __shared__ __align__(16) float w2_s[LAT];
  __shared__ __align__(16) float w_s[WAVES][CHUNK][WPAD];  // 34 KB, per-lane column
  __shared__ __align__(16) _Float16 P_s[WAVES][QB][CHUNK]; // 4 KB, wave-private
  __shared__ float m_s[WAVES][QB];
  __shared__ float l_s[WAVES][QB];
  __shared__ float acc_s[WAVES][QB][LAT];                  // 16 KB

  const int tid  = threadIdx.x;
  const int wave = tid >> 5;
  const int lane = tid & 31;
  const int hi   = lane >> 4;     // which 16-lane half
  const int lo   = lane & 15;
  const int qb   = blockIdx.x * QB;

  for (int i = tid; i < QB * LAT; i += 128)
    u_s[i >> 6][i & 63] = u[(qb + (i >> 6)) * LAT + (i & 63)];
  if (tid < QB * 3) posq_s[tid / 3][tid % 3] = posq[(qb + tid / 3) * 3 + tid % 3];
  if (tid < LAT) w2_s[tid] = W2[tid];
  __syncthreads();

  const float b2v = b2[0];
  const v2f z2 = {0.f, 0.f};

  // named per-query state: sv (packed logit accum), s, mrun (online max), sc
#define DECLQ(q) v2f sv##q; float s##q, mrun##q = NEGINF, sc##q = 1.f;
  FOREACH_Q(DECLQ)

  v8f acc[4], accl;                 // P@V tiles + P@ones (denominator) tile
#pragma unroll
  for (int r = 0; r < 8; ++r) {
#pragma unroll
    for (int t = 0; t < 4; ++t) acc[t][r] = 0.f;
    accl[r] = 0.f;
  }

  // B "ones" tile: column N=0 (lanes with lo==0) all-ones, else zero.
  const _Float16 onev = (_Float16)((lo == 0) ? 1.0f : 0.0f);
  v16h Bones;
#pragma unroll
  for (int i = 0; i < 16; ++i) Bones[i] = onev;

  const int obase0 = wave * OBS_PER_WAVE;
  float* const wcol = &w_s[wave][lane][0];   // this lane's private LDS column

  for (int c = 0; c < NCHUNK; ++c) {
    const int o = obase0 + c * CHUNK + lane;

    // stage this lane's obs column w[o][0..63] into padded LDS (same-thread use)
    const float4* wp = (const float4*)(w + (size_t)o * LAT);
#pragma unroll
    for (int i = 0; i < 16; ++i)
      *(float4*)(wcol + i * 4) = wp[i];
    if (c + 1 < NCHUNK) __builtin_prefetch(w + (size_t)(o + CHUNK) * LAT, 0, 0);
    const float ox = poso[o * 3 + 0], oy = poso[o * 3 + 1], oz = poso[o * 3 + 2];

    // logits: s[q] = b2 + sum_l relu(u[q][l] + w[o][l]) * W2[l]
    // packed-f32 pipeline: v_pk_add (u+w), v_max pair (relu), v_pk_fma (dot W2)
#define SINIT(q) sv##q = z2;
    FOREACH_Q(SINIT)
    for (int l4 = 0; l4 < 16; ++l4) {          // rolled: all operands LDS-addressed
      const float4 wv  = *(const float4*)(wcol + l4 * 4);
      const float4 w2v = *(const float4*)&w2_s[l4 * 4];
      const v2f w01 = {wv.x, wv.y},  w23 = {wv.z, wv.w};
      const v2f q01 = {w2v.x, w2v.y}, q23 = {w2v.z, w2v.w};
#define LOGQ(q) { const float4 uv = *(const float4*)&u_s[q][l4 * 4];            \
      v2f a0 = {uv.x, uv.y}; v2f a1 = {uv.z, uv.w};                             \
      a0 += w01; a1 += w23;                                                     \
      a0 = __builtin_elementwise_max(a0, z2);                                   \
      a1 = __builtin_elementwise_max(a1, z2);                                   \
      sv##q = a0 * q01 + sv##q;                                                 \
      sv##q = a1 * q23 + sv##q; }
      FOREACH_Q(LOGQ)
    }
#define SFIN(q) s##q = sv##q.x + sv##q.y + b2v;
    FOREACH_Q(SFIN)

    // P_s is wave-private; LDS ops within a wave are hardware in-order (DScnt),
    // so only a compiler scheduling fence is needed around store->load reuse.
    __builtin_amdgcn_wave_barrier();

    // mask + online max (per query row; obs striped over 32 lanes)
#define SOFTQ(q) {                                                              \
      const float dx = posq_s[q][0] - ox;                                       \
      const float dy = posq_s[q][1] - oy;                                       \
      const float dz = posq_s[q][2] - oz;                                       \
      const float d2 = fmaf(dx, dx, fmaf(dy, dy, dz * dz));                     \
      const bool msk = (d2 > RADIUS2);                                          \
      const float sq = msk ? NEGINF : s##q;                                     \
      float rmax = sq;                                                          \
      rmax = fmaxf(rmax, __shfl_xor(rmax, 16, 32));                             \
      rmax = fmaxf(rmax, __shfl_xor(rmax, 8, 32));                              \
      rmax = fmaxf(rmax, __shfl_xor(rmax, 4, 32));                              \
      rmax = fmaxf(rmax, __shfl_xor(rmax, 2, 32));                              \
      rmax = fmaxf(rmax, __shfl_xor(rmax, 1, 32));                              \
      const float mnew = fmaxf(mrun##q, rmax);                                  \
      const float p = msk ? 0.f : __expf(sq - mnew);                            \
      sc##q = (mnew == NEGINF) ? 1.f : __expf(mrun##q - mnew);                  \
      mrun##q = mnew;                                                           \
      P_s[wave][q][lane] = (_Float16)p; }
    FOREACH_Q(SOFTQ)
    __builtin_amdgcn_wave_barrier();

    // rescale accumulators per row (C layout: VGPR r -> row r + 8*hi)
    const bool loh = (lane < 16);
    const float f0 = loh ? sc0 : sc8;
    const float f1 = loh ? sc1 : sc9;
    const float f2 = loh ? sc2 : sc10;
    const float f3 = loh ? sc3 : sc11;
    const float f4 = loh ? sc4 : sc12;
    const float f5 = loh ? sc5 : sc13;
    const float f6 = loh ? sc6 : sc14;
    const float f7 = loh ? sc7 : sc15;
#define RESCT(T) { T[0] *= f0; T[1] *= f1; T[2] *= f2; T[3] *= f3;              \
                   T[4] *= f4; T[5] *= f5; T[6] *= f6; T[7] *= f7; }
    RESCT(acc[0]) RESCT(acc[1]) RESCT(acc[2]) RESCT(acc[3]) RESCT(accl)

    // A operand: lane lo, halves {0..7,16..23} (lanes 0-15) / {8..15,24..31} (16-31)
    const _Float16* prow = &P_s[wave][lo][hi ? 8 : 0];
    v16h A;
    ((v8h*)&A)[0] = *(const v8h*)(prow);
    ((v8h*)&A)[1] = *(const v8h*)(prow + 16);

    // denominator tile: accl += P @ ones (column 0 = running row-sum)
    accl = __builtin_amdgcn_wmma_f32_16x16x32_f16(
        false, A, false, Bones, (short)0, accl, false, false);

    // B operands from vT: lanes 0-15 K{0..7,16..23}, lanes 16-31 K{8..15,24..31}
    const int kbase = obase0 + c * CHUNK + hi * 8;
#pragma unroll
    for (int t = 0; t < 4; ++t) {
      const _Float16* bp = vT + (size_t)(t * 16 + lo) * NO + kbase;
      v16h B;
      ((v8h*)&B)[0] = *(const v8h*)(bp);
      ((v8h*)&B)[1] = *(const v8h*)(bp + 16);
      acc[t] = __builtin_amdgcn_wmma_f32_16x16x32_f16(
          false, A, false, B, (short)0, acc[t], false, false);
    }
  }

  // spill per-wave partials for cross-wave flash merge.
  // m: replicated across lanes -> lane 0, constant indices.
  if (lane == 0) {
#define STOREM(q) m_s[wave][q] = mrun##q;
    FOREACH_Q(STOREM)
  }
  // l: column 0 of the P@ones tile -> lanes with lo==0 hold rows r+8*hi.
  if (lo == 0) {
#pragma unroll
    for (int r = 0; r < 8; ++r) l_s[wave][r + 8 * hi] = accl[r];
  }
#pragma unroll
  for (int t = 0; t < 4; ++t)
#pragma unroll
    for (int r = 0; r < 8; ++r)
      acc_s[wave][r + 8 * hi][t * 16 + lo] = acc[t][r];
  __syncthreads();

  // merge 4 waves and normalize
  for (int idx = tid; idx < QB * LAT; idx += 128) {
    const int row = idx >> 6, col = idx & 63;
    float mm = NEGINF;
#pragma unroll
    for (int wv = 0; wv < WAVES; ++wv) mm = fmaxf(mm, m_s[wv][row]);
    float den = 0.f, num = 0.f;
#pragma unroll
    for (int wv = 0; wv < WAVES; ++wv) {
      const float f = (m_s[wv][row] == NEGINF) ? 0.f : __expf(m_s[wv][row] - mm);
      den = fmaf(f, l_s[wv][row], den);
      num = fmaf(f, acc_s[wv][row][col], num);
    }
    out[(size_t)(qb + row) * LAT + col] = (den > 0.f) ? (num / den) : 0.f;
  }
}

// ---------------------------------------------------------------------------
extern "C" void kernel_launch(void* const* d_in, const int* in_sizes, int n_in,
                              void* d_out, int out_size, void* d_ws, size_t ws_size,
                              hipStream_t stream) {
  const float* h_obs     = (const float*)d_in[0];
  // d_in[1] = x_obs (unused by the reference output)
  const float* pos_obs   = (const float*)d_in[2];
  const float* pos_query = (const float*)d_in[3];
  const float* W1        = (const float*)d_in[4];
  const float* b1        = (const float*)d_in[5];
  const float* W2        = (const float*)d_in[6];
  const float* b2        = (const float*)d_in[7];
  const float* Wv        = (const float*)d_in[8];
  const float* bv        = (const float*)d_in[9];
  float* out = (float*)d_out;

  char* ws = (char*)d_ws;
  float*     u  = (float*)(ws);                         // 512 KB
  float*     w  = (float*)(ws + 512 * 1024);            // 512 KB
  _Float16*  vT = (_Float16*)(ws + 1024 * 1024);        // 256 KB

  {
    const int total = 2 * NQ * LAT;
    gano_prep_uw<<<(total + 255) / 256, 256, 0, stream>>>(pos_query, pos_obs, W1, b1, u, w);
  }
  {
    const int total = NO * LAT;
    gano_prep_v<<<(total + 255) / 256, 256, 0, stream>>>(h_obs, Wv, bv, vT);
  }
  gano_attn<<<NQ / QB, WAVES * 32, 0, stream>>>(u, w, vT, pos_query, pos_obs, W2, b2, out);
}
